// CausalSelfAttention_30262339567865
// MI455X (gfx1250) — matrix operand order repaired and checked by reading, not compiled, not run on hardware
//
#include <hip/hip_runtime.h>

#define DEVI __device__ __forceinline__

typedef __bf16 bf16_t;
typedef bf16_t v16bf __attribute__((ext_vector_type(16)));
typedef float  v8f   __attribute__((ext_vector_type(8)));

namespace {
constexpr int B_  = 4;
constexpr int T_  = 2048;
constexpr int C_  = 1024;
constexpr int H_  = 16;
constexpr int D_  = 64;
constexpr int M_  = B_ * T_;          // 8192 token rows
constexpr float LN1E4_OVER_32 = 0.28782313662425574f;  // ln(10000)/32
}

// ---------- helpers ----------

DEVI unsigned short f2bf(float f) {
    unsigned int u = __float_as_uint(f);
    u += 0x7FFFu + ((u >> 16) & 1u);   // round-to-nearest-even
    return (unsigned short)(u >> 16);
}

union ABu { unsigned int u[8]; v16bf v; };

// A operand (16x32 bf16) from row-major [row][k], stride ld (elems).
DEVI v16bf load_A_rm(const unsigned short* base, int ld) {
    const int lane = threadIdx.x & 31;
    const int row  = lane & 15;
    const int kofs = (lane >> 4) << 4;            // 0 or 16
    ABu t;
    const unsigned short* p0 = base + row * ld + kofs;
    const unsigned short* p1 = p0 + 8;
#pragma unroll
    for (int i = 0; i < 4; ++i) {
        t.u[i]     = *(const unsigned int*)(p0 + 2 * i);
        t.u[i + 4] = *(const unsigned int*)(p1 + 2 * i);
    }
    return t.v;
}

// B operand (32x16 bf16) from row-major [n][k] source, stride ld.
DEVI v16bf load_B_nk(const unsigned short* base, int ld) {
    const int lane = threadIdx.x & 31;
    const int n    = lane & 15;
    const int kb   = (lane >> 4) << 4;            // 0 or 16
    ABu t;
    const unsigned short* p = base + n * ld + kb;
#pragma unroll
    for (int i = 0; i < 8; ++i) t.u[i] = *(const unsigned int*)(p + 2 * i);
    return t.v;
}

DEVI v8f wmma_bf16(v16bf a, v16bf b, v8f c) {
    return __builtin_amdgcn_wmma_f32_16x16x32_bf16(
        false, a, false, b, (short)0, c, false, false);
}

// 32x64 output tile per wave: 2 A sub-tiles share 4 B sub-tiles -> 8 WMMA / K-step.
// All operands loaded into distinct registers first so loads clause together
// and the loadcnt waits amortize over the whole WMMA group.
DEVI void gemm_tile_32x64(const unsigned short* __restrict__ A, int lda,
                          const unsigned short* __restrict__ W, int ldw,
                          int K, v8f acc[2][4]) {
    for (int k0 = 0; k0 < K; k0 += 32) {
        v16bf a0 = load_A_rm(A + k0, lda);
        v16bf a1 = load_A_rm(A + 16 * lda + k0, lda);
        v16bf b0 = load_B_nk(W + 0 * 16 * ldw + k0, ldw);
        v16bf b1 = load_B_nk(W + 1 * 16 * ldw + k0, ldw);
        v16bf b2 = load_B_nk(W + 2 * 16 * ldw + k0, ldw);
        v16bf b3 = load_B_nk(W + 3 * 16 * ldw + k0, ldw);
        acc[0][0] = wmma_bf16(a0, b0, acc[0][0]);
        acc[0][1] = wmma_bf16(a0, b1, acc[0][1]);
        acc[0][2] = wmma_bf16(a0, b2, acc[0][2]);
        acc[0][3] = wmma_bf16(a0, b3, acc[0][3]);
        acc[1][0] = wmma_bf16(a1, b0, acc[1][0]);
        acc[1][1] = wmma_bf16(a1, b1, acc[1][1]);
        acc[1][2] = wmma_bf16(a1, b2, acc[1][2]);
        acc[1][3] = wmma_bf16(a1, b3, acc[1][3]);
    }
}

// ---------- kernel 1: fp32 -> bf16 convert ----------

__global__ void k_cvt_bf16(const float* __restrict__ in,
                           unsigned short* __restrict__ out, int n) {
    int i = blockIdx.x * blockDim.x + threadIdx.x;
    int stride = gridDim.x * blockDim.x;
    for (; i < n; i += stride) out[i] = f2bf(in[i]);
}

// ---------- kernel 2: QKV GEMM (bf16 WMMA) with fused RoPE epilogue ----------
// Each wave owns a 32x64 tile; 64-wide N aligned to one head.
// Writes Q/K [B,H,T,D] bf16 and V transposed [B,H,D,T] bf16.

__global__ __launch_bounds__(256) void k_qkv_rope(
        const unsigned short* __restrict__ xb,
        const unsigned short* __restrict__ wb,
        unsigned short* __restrict__ Qb,
        unsigned short* __restrict__ Kb,
        unsigned short* __restrict__ Vt) {
    const int lane = threadIdx.x & 31;
    const int wid  = blockIdx.x * (blockDim.x >> 5) + (threadIdx.x >> 5);
    const int NT   = (3 * C_) / 64;               // 48 N-tiles
    const int mt   = wid / NT, nt = wid % NT;
    const int m0   = mt << 5, n0 = nt << 6;

    v8f acc[2][4] = {{v8f(0.f), v8f(0.f), v8f(0.f), v8f(0.f)},
                     {v8f(0.f), v8f(0.f), v8f(0.f), v8f(0.f)}};
    gemm_tile_32x64(xb + m0 * C_, C_, wb + n0 * C_, C_, C_, acc);

    const int region = n0 / C_;                   // 0=q 1=k 2=v
    const int h      = (n0 % C_) >> 6;            // head
    const int col    = lane & 15;
    const int rowh   = (lane >> 4) << 3;          // 0 or 8

#pragma unroll
    for (int mi = 0; mi < 2; ++mi) {
#pragma unroll
        for (int r = 0; r < 8; ++r) {
            const int m  = m0 + 16 * mi + rowh + r;   // global token row
            const int bi = m >> 11;                   // / T_
            const int tp = m & (T_ - 1);              // token position
            float a0 = acc[mi][0][r], a1 = acc[mi][1][r];
            float a2 = acc[mi][2][r], a3 = acc[mi][3][r];
            if (region < 2) {
                // RoPE: pair (d, d+32) = (subtile j, j+2), same lane.
                float ang0 = (float)tp * __expf(-(float)col        * LN1E4_OVER_32);
                float ang1 = (float)tp * __expf(-(float)(16 + col) * LN1E4_OVER_32);
                float s0, c0, s1, c1;
                __sincosf(ang0, &s0, &c0);
                __sincosf(ang1, &s1, &c1);
                float r0 = a0 * c0 - a2 * s0, r2 = a2 * c0 + a0 * s0;
                float r1 = a1 * c1 - a3 * s1, r3 = a3 * c1 + a1 * s1;
                unsigned short* dst =
                    (region == 0 ? Qb : Kb) + (((bi * H_ + h) * T_ + tp) << 6);
                dst[col]      = f2bf(r0);
                dst[16 + col] = f2bf(r1);
                dst[32 + col] = f2bf(r2);
                dst[48 + col] = f2bf(r3);
            } else {
                unsigned short* dst = Vt + (((bi * H_ + h) << 6) * T_) + tp;
                dst[(col)      * T_] = f2bf(a0);
                dst[(16 + col) * T_] = f2bf(a1);
                dst[(32 + col) * T_] = f2bf(a2);
                dst[(48 + col) * T_] = f2bf(a3);
            }
        }
    }
}

// ---------- kernel 3: flash attention ----------
// One wave per 16-query tile of one (b,h); 32-key tiles, online softmax.
// V B-operands are loaded BEFORE the softmax VALU block so the exp/shuffle
// work hides their latency; P goes through a wave-private LDS bounce to
// convert WMMA C-layout -> A-layout.

__global__ __launch_bounds__(256) void k_flash(
        const unsigned short* __restrict__ Qb,
        const unsigned short* __restrict__ Kb,
        const unsigned short* __restrict__ Vt,
        unsigned short* __restrict__ Yb) {
    __shared__ unsigned short lp[8][16 * 34];     // per-wave 16x32 P tile, padded

    const int lane = threadIdx.x & 31;
    const int wv   = threadIdx.x >> 5;
    const int wid  = blockIdx.x * 8 + wv;
    const int qt   = wid & (T_ / 16 - 1);         // 128 q-tiles
    const int bh   = wid >> 7;
    const int q0   = qt << 4;

    const unsigned short* qp = Qb + (bh * T_ + q0) * D_;
    const unsigned short* kp = Kb + bh * T_ * D_;
    const unsigned short* vp = Vt + bh * D_ * T_;

    const v16bf qa0 = load_A_rm(qp, D_);
    const v16bf qa1 = load_A_rm(qp + 32, D_);

    v8f acc[4] = {v8f(0.f), v8f(0.f), v8f(0.f), v8f(0.f)};
    float mrow[8], lrow[8];
#pragma unroll
    for (int r = 0; r < 8; ++r) { mrow[r] = -__builtin_inff(); lrow[r] = 0.f; }

    const int col  = lane & 15;
    const int rowh = (lane >> 4) << 3;

    for (int kt = 0; kt < q0 + 16; kt += 32) {
        // K operands for the two 16-key halves, both 32-deep d halves.
        v16bf kb00 = load_B_nk(kp + kt * D_,             D_);
        v16bf kb01 = load_B_nk(kp + kt * D_ + 32,        D_);
        v16bf kb10 = load_B_nk(kp + (kt + 16) * D_,      D_);
        v16bf kb11 = load_B_nk(kp + (kt + 16) * D_ + 32, D_);

        // Prefetch next key tile (gfx1250 global_prefetch path).
        if (kt + 32 < q0 + 16) {
            __builtin_prefetch(kp + (kt + 32) * D_, 0, 3);
            __builtin_prefetch(vp + kt + 32, 0, 3);
        }

        v8f s0 = v8f(0.f), s1 = v8f(0.f);
        s0 = wmma_bf16(qa0, kb00, s0);
        s0 = wmma_bf16(qa1, kb01, s0);
        s1 = wmma_bf16(qa0, kb10, s1);
        s1 = wmma_bf16(qa1, kb11, s1);

        // V operands issued early: independent of softmax, latency hidden
        // behind the exp/shuffle VALU work below.
        v16bf vb0 = load_B_nk(vp + 0 * 16 * T_ + kt, T_);
        v16bf vb1 = load_B_nk(vp + 1 * 16 * T_ + kt, T_);
        v16bf vb2 = load_B_nk(vp + 2 * 16 * T_ + kt, T_);
        v16bf vb3 = load_B_nk(vp + 3 * 16 * T_ + kt, T_);

#pragma unroll
        for (int r = 0; r < 8; ++r) {
            const int q = q0 + rowh + r;
            float a = s0[r] * 0.125f;             // 1/sqrt(64)
            float b = s1[r] * 0.125f;
            if (kt + col      > q) a = -__builtin_inff();
            if (kt + 16 + col > q) b = -__builtin_inff();

            float rm = fmaxf(a, b);
#pragma unroll
            for (int x = 1; x < 16; x <<= 1) rm = fmaxf(rm, __shfl_xor(rm, x, 32));
            const float mn = fmaxf(mrow[r], rm);
            const float sc = __expf(mrow[r] - mn);
            const float e0 = __expf(a - mn);
            const float e1 = __expf(b - mn);
            float rs = e0 + e1;
#pragma unroll
            for (int x = 1; x < 16; x <<= 1) rs += __shfl_xor(rs, x, 32);
            lrow[r] = lrow[r] * sc + rs;
            mrow[r] = mn;
#pragma unroll
            for (int t = 0; t < 4; ++t) acc[t][r] *= sc;

            lp[wv][(rowh + r) * 34 + col]      = f2bf(e0);
            lp[wv][(rowh + r) * 34 + col + 16] = f2bf(e1);
        }

        // DS ops are in-order within a wave; stop compiler reordering + drain.
        asm volatile("s_wait_dscnt 0" ::: "memory");
        const v16bf pa = load_A_rm(&lp[wv][0], 34);
        acc[0] = wmma_bf16(pa, vb0, acc[0]);
        acc[1] = wmma_bf16(pa, vb1, acc[1]);
        acc[2] = wmma_bf16(pa, vb2, acc[2]);
        acc[3] = wmma_bf16(pa, vb3, acc[3]);
        asm volatile("" ::: "memory");
    }

    const int bi = bh >> 4;                       // / H_
    const int h  = bh & 15;
#pragma unroll
    for (int r = 0; r < 8; ++r) {
        const float inv = 1.0f / lrow[r];
        const int q = q0 + rowh + r;
        unsigned short* dst = Yb + (bi * T_ + q) * C_ + (h << 6);
#pragma unroll
        for (int t = 0; t < 4; ++t) dst[t * 16 + col] = f2bf(acc[t][r] * inv);
    }
}

// ---------- kernel 4: output projection GEMM (bf16 WMMA, fp32 out) ----------

__global__ __launch_bounds__(256) void k_proj(
        const unsigned short* __restrict__ Yb,
        const unsigned short* __restrict__ Wb,
        float* __restrict__ out) {
    const int lane = threadIdx.x & 31;
    const int wid  = blockIdx.x * (blockDim.x >> 5) + (threadIdx.x >> 5);
    const int NT   = C_ / 64;                     // 16
    const int mt   = wid / NT, nt = wid % NT;
    const int m0   = mt << 5, n0 = nt << 6;

    v8f acc[2][4] = {{v8f(0.f), v8f(0.f), v8f(0.f), v8f(0.f)},
                     {v8f(0.f), v8f(0.f), v8f(0.f), v8f(0.f)}};
    gemm_tile_32x64(Yb + m0 * C_, C_, Wb + n0 * C_, C_, C_, acc);

    const int col  = lane & 15;
    const int rowh = (lane >> 4) << 3;
#pragma unroll
    for (int mi = 0; mi < 2; ++mi) {
#pragma unroll
        for (int r = 0; r < 8; ++r) {
            float* dst = out + (m0 + 16 * mi + rowh + r) * C_ + n0;
#pragma unroll
            for (int j = 0; j < 4; ++j) dst[j * 16 + col] = acc[mi][j][r];
        }
    }
}

// ---------- host orchestration ----------

extern "C" void kernel_launch(void* const* d_in, const int* in_sizes, int n_in,
                              void* d_out, int out_size, void* d_ws, size_t ws_size,
                              hipStream_t stream) {
    const float* x     = (const float*)d_in[0];   // [4,2048,1024]
    const float* Wqkv  = (const float*)d_in[1];   // [3072,1024]
    const float* Wproj = (const float*)d_in[2];   // [1024,1024]
    float* out = (float*)d_out;

    char* ws = (char*)d_ws;
    const size_t MB = 1024 * 1024;
    unsigned short* xb     = (unsigned short*)(ws + 0 * MB);   // 16 MB
    unsigned short* wqkvb  = (unsigned short*)(ws + 16 * MB);  //  6 MB
    unsigned short* wprojb = (unsigned short*)(ws + 24 * MB);  //  2 MB
    unsigned short* Qb     = (unsigned short*)(ws + 32 * MB);  // 16 MB
    unsigned short* Kb     = (unsigned short*)(ws + 48 * MB);  // 16 MB
    unsigned short* Vt     = (unsigned short*)(ws + 64 * MB);  // 16 MB
    unsigned short* Yb     = (unsigned short*)(ws + 80 * MB);  // 16 MB

    const int nx = M_ * C_;            // 8388608
    const int nq = 3 * C_ * C_;        // 3145728
    const int np = C_ * C_;            // 1048576

    k_cvt_bf16<<<2048, 256, 0, stream>>>(x, xb, nx);
    k_cvt_bf16<<<1024, 256, 0, stream>>>(Wqkv, wqkvb, nq);
    k_cvt_bf16<<<512, 256, 0, stream>>>(Wproj, wprojb, np);

    // QKV GEMM+RoPE: (8192/32)*(3072/64) = 12288 waves / 8 per block
    k_qkv_rope<<<1536, 256, 0, stream>>>(xb, wqkvb, Qb, Kb, Vt);

    // Flash attention: 4*16*128 = 8192 waves
    k_flash<<<1024, 256, 0, stream>>>(Qb, Kb, Vt, Yb);

    // Output projection: (8192/32)*(1024/64) = 4096 waves
    k_proj<<<512, 256, 0, stream>>>(Yb, wprojb, out);
}
